// LoRAFirstLayerMLP_31731218383074
// MI455X (gfx1250) — matrix operand order repaired and checked by reading, not compiled, not run on hardware
//
#include <hip/hip_runtime.h>
#include <hip/hip_bf16.h>

typedef __bf16 bf16_t;
typedef __attribute__((ext_vector_type(16))) __bf16 v16bf;
typedef __attribute__((ext_vector_type(8)))  __bf16 v8bf;
typedef __attribute__((ext_vector_type(4)))  __bf16 v4bf;
typedef __attribute__((ext_vector_type(8)))  float  v8f;
typedef __attribute__((ext_vector_type(4)))  float  v4f;

constexpr int BATCH = 16384;
constexpr int DIN   = 1024;
constexpr int MHID  = 4096;
constexpr int RANK  = 16;
constexpr float SCALING = 2.0f;   // alpha/r = 32/16

constexpr int NSPLIT  = 8;                // split of M (hidden) across blocks
constexpr int MCHUNK  = MHID / NSPLIT;    // 512
constexpr int NTILE   = 4;                // m-tiles per wave concurrently
constexpr int RTILE   = 2;                // 16-row batch tiles per wave
constexpr int ROWS_PER_WG = 8 * 16 * RTILE;  // 8 waves * 32 rows = 256

// workspace layout (bytes), all 256-aligned
constexpr size_t WS_WEFF = 0;                                   // MHID*DIN bf16  = 8 MB
constexpr size_t WS_XBF  = (size_t)MHID * DIN * 2;              // BATCH*DIN bf16 = 32 MB
constexpr size_t WS_PART = WS_XBF + (size_t)BATCH * DIN * 2;    // NSPLIT*BATCH f32

// ---------------------------------------------------------------------------
// Kernel 1: W_eff = W0 + 2*(B@A), f32 -> bf16.
// One block per m-row (m = blockIdx.x uniform -> B[m,*] scalarizes).
// ---------------------------------------------------------------------------
__global__ void __launch_bounds__(256)
weff_kernel(const float* __restrict__ W0, const float* __restrict__ A,
            const float* __restrict__ B, bf16_t* __restrict__ weff) {
  const int m = blockIdx.x;              // 0..MHID-1
  const int d = threadIdx.x * 4;         // 0..DIN-1, 4 contiguous
  const size_t idx = (size_t)m * DIN + d;
  v4f w = *(const v4f*)(W0 + idx);
#pragma unroll
  for (int r = 0; r < RANK; ++r) {
    const float bv = SCALING * B[m * RANK + r];   // block-uniform -> s_load
    const float* arow = A + (size_t)r * DIN + d;
    w[0] = fmaf(bv, arow[0], w[0]);
    w[1] = fmaf(bv, arow[1], w[1]);
    w[2] = fmaf(bv, arow[2], w[2]);
    w[3] = fmaf(bv, arow[3], w[3]);
  }
  v4bf o;
  o[0] = (__bf16)w[0]; o[1] = (__bf16)w[1]; o[2] = (__bf16)w[2]; o[3] = (__bf16)w[3];
  *(v4bf*)(weff + idx) = o;
}

// ---------------------------------------------------------------------------
// Kernel 2: x f32 -> bf16
// ---------------------------------------------------------------------------
__global__ void __launch_bounds__(256)
xcvt_kernel(const float* __restrict__ x, bf16_t* __restrict__ xb) {
  const size_t idx = ((size_t)blockIdx.x * 256 + threadIdx.x) * 4;
  v4f v = *(const v4f*)(x + idx);
  v4bf o;
  o[0] = (__bf16)v[0]; o[1] = (__bf16)v[1]; o[2] = (__bf16)v[2]; o[3] = (__bf16)v[3];
  *(v4bf*)(xb + idx) = o;
}

// ---------------------------------------------------------------------------
// Kernel 3: fused GEMM + ReLU + fc2 partial reduction.
//   grid = (BATCH/256) * NSPLIT blocks, 256 threads (8 wave32s).
//   Each wave: 2x16 batch rows x MCHUNK hidden cols, K = 1024.
//   2x4 register blocking: 12 b128 loads feed 8 WMMAs per k-step.
// ---------------------------------------------------------------------------
__global__ void __launch_bounds__(256)
gemm_kernel(const bf16_t* __restrict__ xb, const bf16_t* __restrict__ weff,
            const float* __restrict__ b0, const float* __restrict__ w2,
            float* __restrict__ partial) {
  const int lane = threadIdx.x & 31;
  const int wave = threadIdx.x >> 5;
  const int n  = lane & 15;   // A: M row / B: N col / D: N col
  const int hi = lane >> 4;   // K-half selector (A,B); row-half selector (D)

  const int row_blk  = blockIdx.x & (BATCH / ROWS_PER_WG - 1); // 0..63
  const int msplit   = blockIdx.x / (BATCH / ROWS_PER_WG);     // 0..NSPLIT-1
  const int row_base = row_blk * ROWS_PER_WG + wave * (16 * RTILE);
  const int m_begin  = msplit * MCHUNK;

  // A operand: lane holds x[row+n][k + 16*hi .. +15]
  const bf16_t* xp0 = xb + (size_t)(row_base + n) * DIN + 16 * hi;
  const bf16_t* xp1 = xp0 + (size_t)16 * DIN;

  float sums[RTILE][8];
#pragma unroll
  for (int rt = 0; rt < RTILE; ++rt)
#pragma unroll
    for (int j = 0; j < 8; ++j) sums[rt][j] = 0.f;

  for (int mg = 0; mg < MCHUNK / (16 * NTILE); ++mg) {   // 8 groups of 4 m-tiles
    const int m_base0 = m_begin + mg * 16 * NTILE;

    // B operand ptrs: lane holds W_eff[m_base+n][k + 16*hi .. +15]
    const bf16_t* wp[NTILE];
#pragma unroll
    for (int t = 0; t < NTILE; ++t)
      wp[t] = weff + (size_t)(m_base0 + 16 * t + n) * DIN + 16 * hi;

    v8f acc[RTILE][NTILE];
#pragma unroll
    for (int rt = 0; rt < RTILE; ++rt)
#pragma unroll
      for (int t = 0; t < NTILE; ++t) acc[rt][t] = (v8f){};

#pragma unroll 2
    for (int k = 0; k < DIN; k += 32) {
      v8bf a0lo = *(const v8bf*)(xp0 + k);
      v8bf a0hi = *(const v8bf*)(xp0 + k + 8);
      v8bf a1lo = *(const v8bf*)(xp1 + k);
      v8bf a1hi = *(const v8bf*)(xp1 + k + 8);
      v16bf a0 = __builtin_shufflevector(a0lo, a0hi,
                   0, 1, 2, 3, 4, 5, 6, 7, 8, 9, 10, 11, 12, 13, 14, 15);
      v16bf a1 = __builtin_shufflevector(a1lo, a1hi,
                   0, 1, 2, 3, 4, 5, 6, 7, 8, 9, 10, 11, 12, 13, 14, 15);
#pragma unroll
      for (int t = 0; t < NTILE; ++t) {
        v16bf bb = *(const v16bf*)(wp[t] + k);   // shared by both row tiles
        acc[0][t] = __builtin_amdgcn_wmma_f32_16x16x32_bf16(
                        false, a0, false, bb, (short)0, acc[0][t], false, false);
        acc[1][t] = __builtin_amdgcn_wmma_f32_16x16x32_bf16(
                        false, a1, false, bb, (short)0, acc[1][t], false, false);
      }
    }

    // Epilogue: h = relu(acc + b0[m]); sums += h * W2[m]
#pragma unroll
    for (int t = 0; t < NTILE; ++t) {
      const int mcol = m_base0 + 16 * t + n;
      const float bias = b0[mcol];
      const float wv   = w2[mcol];
#pragma unroll
      for (int rt = 0; rt < RTILE; ++rt) {
#pragma unroll
        for (int j = 0; j < 8; ++j) {
          float h = acc[rt][t][j] + bias;
          h = fmaxf(h, 0.0f);
          sums[rt][j] = fmaf(h, wv, sums[rt][j]);
        }
      }
    }
  }

  // Reduce across the 16 N-lanes sharing each batch row (stays within halves).
#pragma unroll
  for (int rt = 0; rt < RTILE; ++rt) {
#pragma unroll
    for (int j = 0; j < 8; ++j) {
      float s = sums[rt][j];
      s += __shfl_xor(s, 1, 32);
      s += __shfl_xor(s, 2, 32);
      s += __shfl_xor(s, 4, 32);
      s += __shfl_xor(s, 8, 32);
      sums[rt][j] = s;
    }
  }
  if (n == 0) {
    // D layout: VGPR j <-> batch row (j + 8*hi); lanes 0 and 16 write.
#pragma unroll
    for (int rt = 0; rt < RTILE; ++rt)
#pragma unroll
      for (int j = 0; j < 8; ++j)
        partial[(size_t)msplit * BATCH + row_base + 16 * rt + 8 * hi + j] =
            sums[rt][j];
  }
}

// ---------------------------------------------------------------------------
// Kernel 4: out[b] = b2 + sum_p partial[p][b]
// ---------------------------------------------------------------------------
__global__ void __launch_bounds__(256)
final_kernel(const float* __restrict__ partial, const float* __restrict__ b2,
             float* __restrict__ out) {
  const int b = blockIdx.x * 256 + threadIdx.x;
  float s = b2[0];
#pragma unroll
  for (int p = 0; p < NSPLIT; ++p) s += partial[(size_t)p * BATCH + b];
  out[b] = s;
}

extern "C" void kernel_launch(void* const* d_in, const int* in_sizes, int n_in,
                              void* d_out, int out_size, void* d_ws, size_t ws_size,
                              hipStream_t stream) {
  (void)in_sizes; (void)n_in; (void)out_size; (void)ws_size;
  const float* x  = (const float*)d_in[0];
  const float* W0 = (const float*)d_in[1];
  const float* b0 = (const float*)d_in[2];
  const float* A  = (const float*)d_in[3];
  const float* B  = (const float*)d_in[4];
  const float* W2 = (const float*)d_in[5];
  const float* b2 = (const float*)d_in[6];
  float* out = (float*)d_out;

  char* ws = (char*)d_ws;
  bf16_t* weff = (bf16_t*)(ws + WS_WEFF);
  bf16_t* xbf  = (bf16_t*)(ws + WS_XBF);
  float*  part = (float*)(ws + WS_PART);

  weff_kernel<<<MHID, 256, 0, stream>>>(W0, A, B, weff);
  xcvt_kernel<<<(BATCH * DIN / 4) / 256, 256, 0, stream>>>(x, xbf);
  gemm_kernel<<<(BATCH / ROWS_PER_WG) * NSPLIT, 256, 0, stream>>>(xbf, weff, b0, W2, part);
  final_kernel<<<BATCH / 256, 256, 0, stream>>>(part, b2, out);
}
